// CausalMultiHeadSelfAttention_18305150615601
// MI455X (gfx1250) — compile-verified, matrix operand order checked by reading, not run on hardware
//
#include <hip/hip_runtime.h>
#include <hip/hip_bf16.h>

typedef __attribute__((ext_vector_type(16))) __bf16 v16bf;
typedef __attribute__((ext_vector_type(8)))  __bf16 v8bf;
typedef __attribute__((ext_vector_type(8)))  float  v8f;

#define TOK_T  2048
#define DMODEL 1024
#define NHEADS 16
#define DK     64
// ln(10000)
#define LN_THETA 9.2103403719761836f

__device__ __forceinline__ v8f wmma_bf16(v16bf a, v16bf b, v8f c) {
    // D = A(16x32 bf16) * B(32x16 bf16) + C(16x16 f32)
    return __builtin_amdgcn_wmma_f32_16x16x32_bf16(
        /*neg_a=*/false, a, /*neg_b=*/false, b,
        /*c_mod=*/(short)0, c, /*reuse_a=*/false, /*reuse_b=*/false);
}

// Load a 16-element bf16 WMMA operand: two contiguous 8-element (16B) chunks,
// 16 elements apart.  Works for both global and LDS pointers.
// Lane layout: lanes 0-15 hold K = 0..7 / 16..23, lanes 16-31 hold 8..15 / 24..31,
// so p must point at (row, kbase + half*8).
__device__ __forceinline__ v16bf load_bf16_operand(const __bf16* p) {
    v8bf c0 = *(const v8bf*)(p);
    v8bf c1 = *(const v8bf*)(p + 16);
    return __builtin_shufflevector(c0, c1,
        0, 1, 2, 3, 4, 5, 6, 7, 8, 9, 10, 11, 12, 13, 14, 15);
}

// Async DMA of a 64(rows) x 32(k) bf16 tile (4KB) from global to LDS,
// issued cooperatively by 128 threads (2 x 16B per thread), tracked by ASYNCcnt.
__device__ __forceinline__ void async_copy_btile(const __bf16* __restrict__ Wn0, // & W[n0][0]
                                                 int k0,
                                                 __bf16* dst,                    // LDS, row-major [64][32]
                                                 int flat)
{
    #pragma unroll
    for (int c = 0; c < 2; ++c) {
        const int off16 = flat + c * 128;        // 16-byte chunk id, 0..255
        const int row   = off16 >> 2;            // 4 chunks (64B) per row
        const int col8  = off16 & 3;             // 8-element chunk within row
        const __bf16* g = Wn0 + (size_t)row * DMODEL + k0 + col8 * 8;
        const unsigned lds = (unsigned)(uintptr_t)(dst + row * 32 + col8 * 8);
        asm volatile("global_load_async_to_lds_b128 %0, %1, off"
                     :: "v"(lds), "v"(g) : "memory");
    }
}

__device__ __forceinline__ void wait_async_all() {
    asm volatile("s_wait_asynccnt 0" ::: "memory");
}

// ---------------------------------------------------------------------------
// Kernel 0: fp32 -> bf16 pack (X and the four weight matrices), 8 elems/thread
// ---------------------------------------------------------------------------
__global__ __launch_bounds__(256)
void pack_bf16_kernel(const float* __restrict__ src, __bf16* __restrict__ dst, int n8) {
    const int i = blockIdx.x * 256 + threadIdx.x;
    if (i >= n8) return;
    const float4* s = (const float4*)src + (size_t)i * 2;
    float4 a = s[0], b = s[1];
    v8bf o;
    o[0] = (__bf16)a.x; o[1] = (__bf16)a.y; o[2] = (__bf16)a.z; o[3] = (__bf16)a.w;
    o[4] = (__bf16)b.x; o[5] = (__bf16)b.y; o[6] = (__bf16)b.z; o[7] = (__bf16)b.w;
    ((v8bf*)dst)[i] = o;
}

// ---------------------------------------------------------------------------
// Kernel 1: Y = X @ W^T (bf16 in, bf16 out), optional fused RoPE.
//   mode 0: Q/K -> RoPE, store [B, h, T, d_k]
//   mode 1: V   -> no RoPE, store transposed [B, h, d_k, T]
// Block = 4 waves computing a 64x64 tile.  The 64x32 B tile is shared by all
// waves: async-DMA'd into double-buffered LDS, overlapped with WMMA compute.
// ---------------------------------------------------------------------------
__global__ __launch_bounds__(128)
void qkv_proj_kernel(const __bf16* __restrict__ X,      // [B*T, D] bf16
                     const __bf16* __restrict__ W,      // [D, D]   bf16
                     const int*    __restrict__ tokpos, // [T]
                     __bf16*       __restrict__ Y,
                     int do_rope, int store_vt)
{
    __shared__ alignas(16) __bf16 btile[2][64][32];   // double-buffered B tile

    const int flat = threadIdx.x;
    const int lane = threadIdx.x & 31;
    const int wave = threadIdx.x >> 5;
    const int half = lane >> 4;
    const int mrow = lane & 15;
    const int m0 = blockIdx.y * 64 + wave * 16;
    const int n0 = blockIdx.x * 64;

    v8f acc[4];
    #pragma unroll
    for (int t = 0; t < 4; ++t)
        acc[t] = (v8f){0.f, 0.f, 0.f, 0.f, 0.f, 0.f, 0.f, 0.f};

    const __bf16* xrow = X + (size_t)(m0 + mrow) * DMODEL + half * 8;
    const __bf16* Wn0  = W + (size_t)n0 * DMODEL;

    // prime buffer 0
    async_copy_btile(Wn0, 0, &btile[0][0][0], flat);
    wait_async_all();
    __syncthreads();

    for (int k0 = 0; k0 < DMODEL; k0 += 32) {
        const int buf = (k0 >> 5) & 1;
        if (k0 + 32 < DMODEL)                      // start DMA of next tile
            async_copy_btile(Wn0, k0 + 32, &btile[buf ^ 1][0][0], flat);

        v16bf a = load_bf16_operand(xrow + k0);    // per-wave A rows (global)
        #pragma unroll
        for (int t = 0; t < 4; ++t) {
            v16bf b = load_bf16_operand(&btile[buf][t * 16 + mrow][half * 8]);
            acc[t] = wmma_bf16(a, b, acc[t]);
        }

        wait_async_all();                          // own async stores landed
        __syncthreads();                           // everyone's landed / done reading
    }

    const int batch = m0 / TOK_T;
    const int trow0 = m0 % TOK_T;

    #pragma unroll
    for (int t = 0; t < 4; ++t) {
        const int ncol = n0 + t * 16 + mrow;
        const int head = ncol >> 6;
        const int idx  = ncol & (DK - 1);
        float vals[8];
        #pragma unroll
        for (int r = 0; r < 8; ++r) vals[r] = acc[t][r];

        if (do_rope) {
            const int p2 = idx >> 1;                    // rotary pair index
            const float inv_freq = __expf(-(float)(2 * p2) * (LN_THETA / (float)DK));
            const bool is_even = (idx & 1) == 0;
            #pragma unroll
            for (int r = 0; r < 8; ++r) {
                const int tok = trow0 + r + half * 8;
                const float pos = (float)tokpos[tok];
                float s, c;
                __sincosf(pos * inv_freq, &s, &c);
                // even/odd RoPE pair lives in the adjacent lane (same register)
                const float part = __shfl_xor(vals[r], 1, 32);
                vals[r] = is_even ? (vals[r] * c - part * s)
                                  : (part * s + vals[r] * c);
            }
        }

        if (store_vt) {
            // transposed: [B, h, d_k, T] for contiguous P@V operand reads
            __bf16* dst = Y + (((size_t)batch * NHEADS + head) * DK + idx) * TOK_T + trow0;
            #pragma unroll
            for (int r = 0; r < 8; ++r)
                dst[r + half * 8] = (__bf16)vals[r];
        } else {
            #pragma unroll
            for (int r = 0; r < 8; ++r) {
                const int tok = trow0 + r + half * 8;
                Y[(((size_t)batch * NHEADS + head) * TOK_T + tok) * DK + idx] = (__bf16)vals[r];
            }
        }
    }
}

// ---------------------------------------------------------------------------
// Kernel 2: causal flash attention per (b,h).
// Q,K in [B*h, T, 64] bf16; V transposed in [B*h, 64, T] bf16.
// Block = 4 waves; each wave owns 16 query rows, loops KV in 32-wide blocks.
// QK^T and P@V via bf16 WMMA; P transposed C->A layout via per-wave LDS.
// ---------------------------------------------------------------------------
__global__ __launch_bounds__(128)
void flash_attn_kernel(const __bf16* __restrict__ Q,
                       const __bf16* __restrict__ K,
                       const __bf16* __restrict__ VT,
                       __bf16*       __restrict__ H)    // [B*T, D] bf16
{
    __shared__ alignas(16) __bf16 pstage[4][16][32];

    const int lane = threadIdx.x & 31;
    const int wave = threadIdx.x >> 5;
    const int half = lane >> 4;
    const int mrow = lane & 15;
    const int bh = blockIdx.y;                       // 0 .. B*h-1
    const int q0 = blockIdx.x * 64 + wave * 16;      // first query row of this wave

    const __bf16* Qb  = Q  + (size_t)bh * TOK_T * DK;
    const __bf16* Kb  = K  + (size_t)bh * TOK_T * DK;
    const __bf16* VTb = VT + (size_t)bh * DK * TOK_T;

    // Q operand (16 rows x 64 d), unscaled; 1/sqrt(d_k) applied post-WMMA
    v16bf aq[2];
    {
        const __bf16* qp = Qb + (size_t)(q0 + mrow) * DK + half * 8;
        aq[0] = load_bf16_operand(qp);        // d = 0..31
        aq[1] = load_bf16_operand(qp + 32);   // d = 32..63
    }

    v8f o[4];
    #pragma unroll
    for (int t = 0; t < 4; ++t)
        o[t] = (v8f){0.f, 0.f, 0.f, 0.f, 0.f, 0.f, 0.f, 0.f};
    float mrun[8], lrun[8];
    #pragma unroll
    for (int r = 0; r < 8; ++r) { mrun[r] = -1e30f; lrun[r] = 0.f; }

    const int jend = q0 + 15;                        // last KV row needed (causal)
    for (int j0 = 0; j0 <= jend; j0 += 32) {
        // ---- S = Q K^T for 32 KV columns (two 16x16 tiles) ----
        float s[2][8];
        #pragma unroll
        for (int nt = 0; nt < 2; ++nt) {
            int krow = j0 + nt * 16 + mrow;
            krow = krow < (TOK_T - 1) ? krow : (TOK_T - 1);   // OOB rows masked below
            const __bf16* kp = Kb + (size_t)krow * DK + half * 8;
            v16bf bk0 = load_bf16_operand(kp);
            v16bf bk1 = load_bf16_operand(kp + 32);
            v8f sa = (v8f){0.f, 0.f, 0.f, 0.f, 0.f, 0.f, 0.f, 0.f};
            sa = wmma_bf16(aq[0], bk0, sa);
            sa = wmma_bf16(aq[1], bk1, sa);
            const int col = j0 + nt * 16 + mrow;
            #pragma unroll
            for (int r = 0; r < 8; ++r) {
                const int row = q0 + r + half * 8;
                s[nt][r] = (col <= row) ? sa[r] * 0.125f : -1e30f;
            }
        }

        // ---- online softmax update (stats replicated across 16-lane halves) ----
        float p0v[8], p1v[8];
        #pragma unroll
        for (int r = 0; r < 8; ++r) {
            float v = fmaxf(s[0][r], s[1][r]);
            #pragma unroll
            for (int off = 8; off > 0; off >>= 1)
                v = fmaxf(v, __shfl_xor(v, off, 32));
            const float mnew  = fmaxf(mrun[r], v);
            const float scale = __expf(mrun[r] - mnew);
            const float p0 = __expf(s[0][r] - mnew);
            const float p1 = __expf(s[1][r] - mnew);
            float rs = p0 + p1;
            #pragma unroll
            for (int off = 8; off > 0; off >>= 1)
                rs += __shfl_xor(rs, off, 32);
            lrun[r] = lrun[r] * scale + rs;
            mrun[r] = mnew;
            p0v[r] = p0; p1v[r] = p1;
            #pragma unroll
            for (int t = 0; t < 4; ++t)
                o[t][r] *= scale;
        }

        // ---- stage P (C-layout) into LDS, re-read in A-operand layout ----
        #pragma unroll
        for (int r = 0; r < 8; ++r) {
            pstage[wave][r + half * 8][mrow]      = (__bf16)p0v[r];
            pstage[wave][r + half * 8][16 + mrow] = (__bf16)p1v[r];
        }
        const __bf16* pr = &pstage[wave][mrow][half * 8];
        v16bf ap = load_bf16_operand(pr);

        // ---- O += P @ V (four 16x16 tiles over d_k); V is pre-transposed ----
        #pragma unroll
        for (int dt = 0; dt < 4; ++dt) {
            const __bf16* vp = VTb + (size_t)(dt * 16 + mrow) * TOK_T + j0 + half * 8;
            v16bf bv = load_bf16_operand(vp);   // slack reads land in next ws buffer
            o[dt] = wmma_bf16(ap, bv, o[dt]);
        }
    }

    // ---- normalize and scatter heads (bf16) to [B*T, D] row-major ----
    const int batch = bh >> 4;
    const int head  = bh & (NHEADS - 1);
    #pragma unroll
    for (int r = 0; r < 8; ++r) {
        const float inv = 1.0f / lrun[r];
        const int tok = q0 + r + half * 8;
        __bf16* dst = H + ((size_t)batch * TOK_T + tok) * DMODEL + head * DK + mrow;
        #pragma unroll
        for (int dt = 0; dt < 4; ++dt)
            dst[dt * 16] = (__bf16)(o[dt][r] * inv);
    }
}

// ---------------------------------------------------------------------------
// Kernel 3: out = heads @ Wo^T  (bf16 in, fp32 row-major out),
// same async-LDS double-buffered B tile as the projection GEMM.
// ---------------------------------------------------------------------------
__global__ __launch_bounds__(128)
void out_proj_kernel(const __bf16* __restrict__ Hd,    // [B*T, D] bf16
                     const __bf16* __restrict__ Wo,    // [D, D]   bf16
                     float*        __restrict__ Out)   // [B*T, D] fp32
{
    __shared__ alignas(16) __bf16 btile[2][64][32];

    const int flat = threadIdx.x;
    const int lane = threadIdx.x & 31;
    const int wave = threadIdx.x >> 5;
    const int half = lane >> 4;
    const int mrow = lane & 15;
    const int m0 = blockIdx.y * 64 + wave * 16;
    const int n0 = blockIdx.x * 64;

    v8f acc[4];
    #pragma unroll
    for (int t = 0; t < 4; ++t)
        acc[t] = (v8f){0.f, 0.f, 0.f, 0.f, 0.f, 0.f, 0.f, 0.f};

    const __bf16* hrow = Hd + (size_t)(m0 + mrow) * DMODEL + half * 8;
    const __bf16* Wn0  = Wo + (size_t)n0 * DMODEL;

    async_copy_btile(Wn0, 0, &btile[0][0][0], flat);
    wait_async_all();
    __syncthreads();

    for (int k0 = 0; k0 < DMODEL; k0 += 32) {
        const int buf = (k0 >> 5) & 1;
        if (k0 + 32 < DMODEL)
            async_copy_btile(Wn0, k0 + 32, &btile[buf ^ 1][0][0], flat);

        v16bf a = load_bf16_operand(hrow + k0);
        #pragma unroll
        for (int t = 0; t < 4; ++t) {
            v16bf b = load_bf16_operand(&btile[buf][t * 16 + mrow][half * 8]);
            acc[t] = wmma_bf16(a, b, acc[t]);
        }

        wait_async_all();
        __syncthreads();
    }

    #pragma unroll
    for (int t = 0; t < 4; ++t) {
        const int ncol = n0 + t * 16 + mrow;
        #pragma unroll
        for (int r = 0; r < 8; ++r)
            Out[(size_t)(m0 + r + half * 8) * DMODEL + ncol] = acc[t][r];
    }
}

// ---------------------------------------------------------------------------
extern "C" void kernel_launch(void* const* d_in, const int* in_sizes, int n_in,
                              void* d_out, int out_size, void* d_ws, size_t ws_size,
                              hipStream_t stream) {
    const float* X      = (const float*)d_in[0];
    const int*   tokpos = (const int*)  d_in[1];
    const float* Wq     = (const float*)d_in[2];
    const float* Wk     = (const float*)d_in[3];
    const float* Wv     = (const float*)d_in[4];
    const float* Wo     = (const float*)d_in[5];
    float*       out    = (float*)d_out;

    const size_t NX = (size_t)4 * TOK_T * DMODEL;          // 8,388,608
    const size_t NW = (size_t)DMODEL * DMODEL;             // 1,048,576

    __bf16* xb  = (__bf16*)d_ws;
    __bf16* wqb = xb  + NX;
    __bf16* wkb = wqb + NW;
    __bf16* wvb = wkb + NW;
    __bf16* wob = wvb + NW;
    __bf16* qb  = wob + NW;   // [B, h, T, d_k]
    __bf16* kb  = qb  + NX;   // [B, h, T, d_k]
    __bf16* vtb = kb  + NX;   // [B, h, d_k, T]  (hb follows: absorbs slack reads)
    __bf16* hb  = vtb + NX;   // [B*T, D]

    // fp32 -> bf16 packs
    pack_bf16_kernel<<<(int)(NX / 8 / 256), 256, 0, stream>>>(X,  xb,  (int)(NX / 8));
    pack_bf16_kernel<<<(int)(NW / 8 / 256), 256, 0, stream>>>(Wq, wqb, (int)(NW / 8));
    pack_bf16_kernel<<<(int)(NW / 8 / 256), 256, 0, stream>>>(Wk, wkb, (int)(NW / 8));
    pack_bf16_kernel<<<(int)(NW / 8 / 256), 256, 0, stream>>>(Wv, wvb, (int)(NW / 8));
    pack_bf16_kernel<<<(int)(NW / 8 / 256), 256, 0, stream>>>(Wo, wob, (int)(NW / 8));

    dim3 gproj(DMODEL / 64, (4 * TOK_T) / 64);   // (16, 128)
    qkv_proj_kernel<<<gproj, 128, 0, stream>>>(xb, wqb, tokpos, qb,  1, 0);
    qkv_proj_kernel<<<gproj, 128, 0, stream>>>(xb, wkb, tokpos, kb,  1, 0);
    qkv_proj_kernel<<<gproj, 128, 0, stream>>>(xb, wvb, tokpos, vtb, 0, 1);

    dim3 gattn(TOK_T / 64, 4 * NHEADS);          // (32, 64)
    flash_attn_kernel<<<gattn, 128, 0, stream>>>(qb, kb, vtb, hb);

    out_proj_kernel<<<gproj, 128, 0, stream>>>(hb, wob, out);
}
